// ball_ball_update_opt_net_72765335929731
// MI455X (gfx1250) — matrix operand order repaired and checked by reading, not compiled, not run on hardware
//
#include <hip/hip_runtime.h>
#include <hip/hip_bf16.h>

typedef __attribute__((ext_vector_type(16))) _Float16 v16h;
typedef __attribute__((ext_vector_type(8)))  _Float16 v8h;
typedef __attribute__((ext_vector_type(8)))  float    v8f;

#define NF 8
#define NH 200
#define KPAD 224          // 7 K-chunks of 32 (data region)
#define KSTRIDE 232       // row stride in halves: 116 dwords -> conflict-free b128
#define NTILES_H 13       // 208 >= 200
#define MT 4              // M tiles (16 rows) per wave -> 64 rows/wave
#define ROWS_PER_BLOCK 512
#define NBLOCKS 256       // 131072 / 512

// packed tile bases (in units of 512 halves per 32x16 tile)
#define L1_BASE 0         // 13 tiles  (KC=1, NT=13)
#define L2_BASE 13        // 91 tiles  (KC=7, NT=13), order nt*7+kc
#define L3_BASE 104       // 91 tiles
#define L4_BASE 195       // 7 tiles   (KC=7, NT=1)
#define NTILES_TOTAL 202

#define SMEM_ACT_BYTES (ROWS_PER_BLOCK * KSTRIDE * 2)   // 237568
#define SMEM_TOTAL (SMEM_ACT_BYTES + 64*4 + 64*4 + 8*4)

// ---------------------------------------------------------------------------
// Invert 8x8 Q once (Gauss-Jordan with partial pivoting), result -> ws[0..63]
// ---------------------------------------------------------------------------
__global__ void invert_q_kernel(const float* __restrict__ Q, float* __restrict__ Qi) {
  if (threadIdx.x != 0 || blockIdx.x != 0) return;
  float a[8][16];
  for (int i = 0; i < 8; ++i)
    for (int j = 0; j < 8; ++j) { a[i][j] = Q[i*8+j]; a[i][8+j] = (i == j) ? 1.f : 0.f; }
  for (int c = 0; c < 8; ++c) {
    int p = c; float best = fabsf(a[c][c]);
    for (int r = c+1; r < 8; ++r) { float v = fabsf(a[r][c]); if (v > best) { best = v; p = r; } }
    if (p != c) for (int j = 0; j < 16; ++j) { float t = a[c][j]; a[c][j] = a[p][j]; a[p][j] = t; }
    float inv = 1.0f / a[c][c];
    for (int j = 0; j < 16; ++j) a[c][j] *= inv;
    for (int r = 0; r < 8; ++r) if (r != c) {
      float f = a[r][c];
      for (int j = 0; j < 16; ++j) a[r][j] -= f * a[c][j];
    }
  }
  for (int i = 0; i < 8; ++i)
    for (int j = 0; j < 8; ++j) Qi[i*8+j] = a[i][8+j];
}

// ---------------------------------------------------------------------------
// Pack fp32 weights into f16 WMMA B-fragment tiles (32x16 per tile).
// Tile element e: lane = e>>4 (n_local = lane&15), s = e&15, k_local = 8*(lane>>4) + s (+8 if s>=8).
// ---------------------------------------------------------------------------
__global__ void pack_weights_kernel(const float* __restrict__ W1, const float* __restrict__ W2,
                                    const float* __restrict__ W3, const float* __restrict__ W4,
                                    _Float16* __restrict__ wp) {
  int t = blockIdx.x;
  const float* W; int nin, nout, ntile, kchunk;
  if (t < L2_BASE)       { W = W1; nin = NF;  nout = NH; ntile = t;            kchunk = 0; }
  else if (t < L3_BASE)  { int i = t - L2_BASE; W = W2; nin = NH; nout = NH; ntile = i/7; kchunk = i%7; }
  else if (t < L4_BASE)  { int i = t - L3_BASE; W = W3; nin = NH; nout = NH; ntile = i/7; kchunk = i%7; }
  else                   { int i = t - L4_BASE; W = W4; nin = NH; nout = NF; ntile = 0;   kchunk = i;   }
  int e0 = threadIdx.x * 2;
  #pragma unroll
  for (int q = 0; q < 2; ++q) {
    int e = e0 + q;
    int k = kchunk * 32 + ((e >> 8) << 3) + (e & 15) + ((e & 8) ? 8 : 0);
    int n = ntile * 16 + ((e >> 4) & 15);
    float v = (k < nin && n < nout) ? W[n * nin + k] : 0.0f;
    wp[t * 512 + e] = (_Float16)v;
  }
}

// ---------------------------------------------------------------------------
// One MLP layer on a wave's 64 rows (4 M-tiles). All A fragments hoisted to
// VGPRs first (big gfx1250 register file), so LDS is updated IN PLACE and
// each B tile loaded from global (L2-resident) feeds 4 WMMAs.
// ---------------------------------------------------------------------------
template<int KC, int NT, int NOUT, bool RELU, bool LAST>
__device__ __forceinline__ void run_layer(_Float16* sAct,
                                          const _Float16* __restrict__ wp,
                                          const float* __restrict__ bias,
                                          float* __restrict__ outp,
                                          int gRowBase, int mlocal, int lane) {
  const int g = lane >> 4;
  const int m = lane & 15;
  v16h A[MT][KC];
  #pragma unroll
  for (int mt = 0; mt < MT; ++mt) {
    const _Float16* rp = sAct + (mlocal + mt * 16 + m) * KSTRIDE;
    #pragma unroll
    for (int kc = 0; kc < KC; ++kc) {
      v8h lo = *(const v8h*)(rp + kc * 32 + g * 8);        // K = kbase+g*8 .. +7
      v8h hi = *(const v8h*)(rp + kc * 32 + 16 + g * 8);   // K = kbase+16+g*8 .. +7
      A[mt][kc] = __builtin_shufflevector(lo, hi, 0,1,2,3,4,5,6,7,8,9,10,11,12,13,14,15);
    }
  }
  const v8f zc = {};
  for (int nt = 0; nt < NT; ++nt) {
    const int n = nt * 16 + m;
    const float bv = (n < NOUT) ? bias[n] : 0.0f;
    v8f acc[MT];
    const _Float16* tp = wp + (nt * KC) * 512 + lane * 16;
    #pragma unroll
    for (int kc = 0; kc < KC; ++kc) {
      v16h Bf = *(const v16h*)(tp + kc * 512);
      #pragma unroll
      for (int mt = 0; mt < MT; ++mt) {
        acc[mt] = __builtin_amdgcn_wmma_f32_16x16x32_f16(
            false, A[mt][kc], false, Bf, (short)0, (kc == 0) ? zc : acc[mt], false, false);
      }
    }
    #pragma unroll
    for (int mt = 0; mt < MT; ++mt) {
      #pragma unroll
      for (int j = 0; j < 8; ++j) {           // D: M = j + 8*g (+16*mt), N = lane&15
        float v = acc[mt][j] + bv;
        if (RELU) v = fmaxf(v, 0.0f);
        if constexpr (!LAST) {
          sAct[(mlocal + mt * 16 + g * 8 + j) * KSTRIDE + n] = (_Float16)v;
        } else {
          if (n < NOUT) outp[(gRowBase + mt * 16 + g * 8 + j) * NOUT + n] = v;
        }
      }
    }
  }
}

// ---------------------------------------------------------------------------
// Fused: fc0 + analytic equality-QP (via precomputed Qinv) + 4-layer MLP.
// ---------------------------------------------------------------------------
__global__ __launch_bounds__(256) void optnet_kernel(
    const float* __restrict__ x, const float* __restrict__ Qi,
    const _Float16* __restrict__ wp,
    const float* __restrict__ W0, const float* __restrict__ b0,
    const float* __restrict__ b1, const float* __restrict__ b2,
    const float* __restrict__ b3, const float* __restrict__ b4,
    float* __restrict__ out) {
  extern __shared__ char smem[];
  _Float16* sAct = (_Float16*)smem;
  float* sW0 = (float*)(smem + SMEM_ACT_BYTES);
  float* sQ  = sW0 + 64;
  float* sb0 = sQ + 64;

  const int tid = threadIdx.x;
  const int lane = tid & 31;
  const int wave = tid >> 5;
  const int mlocal = wave * (MT * 16);
  const int gRowBase = blockIdx.x * ROWS_PER_BLOCK + mlocal;

  if (tid < 64) sW0[tid] = W0[tid];
  else if (tid < 128) sQ[tid - 64] = Qi[tid - 64];
  if (tid < 8) sb0[tid] = b0[tid];
  __syncthreads();

  // ---- per-thread: rows 2t and 2t+1 (keeps producer==consumer wave) ----
  #pragma unroll
  for (int rr = 0; rr < 2; ++rr) {
    const int lrow = 2 * tid + rr;
    const int row = blockIdx.x * ROWS_PER_BLOCK + lrow;
    _Float16* myrow = sAct + lrow * KSTRIDE;
    v8h zv = {};
    #pragma unroll
    for (int i = 0; i < KSTRIDE; i += 8) *(v8h*)(myrow + i) = zv;   // 232/8 = 29 stores

    float xv[8], h[8], u[8], c0[8], c1[8];
    const float* xr = x + row * NF;
    #pragma unroll
    for (int i = 0; i < 8; ++i) xv[i] = xr[i];
    #pragma unroll
    for (int j = 0; j < 8; ++j) {
      float s = sb0[j];
      #pragma unroll
      for (int i = 0; i < 8; ++i) s += sW0[j * 8 + i] * xv[i];
      h[j] = s;
    }
    const float h3 = h[3];
    #pragma unroll
    for (int i = 0; i < 8; ++i) {
      float s = 0.f;
      #pragma unroll
      for (int j = 0; j < 8; ++j) s += sQ[i * 8 + j] * h[j];
      u[i] = s;
      c0[i] = h3 * sQ[i * 8 + 4] + sQ[i * 8 + 5];
      c1[i] = h3 * sQ[i * 8 + 6] + sQ[i * 8 + 7];
    }
    // M = A Qinv A' (2x2), rhs = A u - b
    const float M00 = h3 * c0[4] + c0[5], M01 = h3 * c1[4] + c1[5];
    const float M10 = h3 * c0[6] + c0[7], M11 = h3 * c1[6] + c1[7];
    const float r0 = (h3 * u[4] + u[5]) - (h3 * h[4] + h[5]);
    const float r1 = (h3 * u[6] + u[7]) - (h3 * h[6] + h[7]);
    const float inv = 1.0f / (M00 * M11 - M01 * M10);
    const float nu0 = (r0 * M11 - r1 * M01) * inv;
    const float nu1 = (M00 * r1 - M10 * r0) * inv;
    #pragma unroll
    for (int i = 0; i < 8; ++i) myrow[i] = (_Float16)(u[i] - nu0 * c0[i] - nu1 * c1[i]);
  }
  // No barrier needed: wave w consumes exactly the 64 rows its own lanes wrote.

  run_layer<1, NTILES_H, NH, true,  false>(sAct, wp + L1_BASE * 512, b1, nullptr, gRowBase, mlocal, lane);
  run_layer<7, NTILES_H, NH, true,  false>(sAct, wp + L2_BASE * 512, b2, nullptr, gRowBase, mlocal, lane);
  run_layer<7, NTILES_H, NH, true,  false>(sAct, wp + L3_BASE * 512, b3, nullptr, gRowBase, mlocal, lane);
  run_layer<7, 1,        NF, false, true >(sAct, wp + L4_BASE * 512, b4, out,     gRowBase, mlocal, lane);
}

extern "C" void kernel_launch(void* const* d_in, const int* in_sizes, int n_in,
                              void* d_out, int out_size, void* d_ws, size_t ws_size,
                              hipStream_t stream) {
  // setup_inputs order: x, Q, W0, b0, W1, b1, W2, b2, W3, b3, W4, b4
  const float* x  = (const float*)d_in[0];
  const float* Q  = (const float*)d_in[1];
  const float* W0 = (const float*)d_in[2];
  const float* b0 = (const float*)d_in[3];
  const float* W1 = (const float*)d_in[4];
  const float* b1 = (const float*)d_in[5];
  const float* W2 = (const float*)d_in[6];
  const float* b2 = (const float*)d_in[7];
  const float* W3 = (const float*)d_in[8];
  const float* b3 = (const float*)d_in[9];
  const float* W4 = (const float*)d_in[10];
  const float* b4 = (const float*)d_in[11];
  float* out = (float*)d_out;

  float* Qi = (float*)d_ws;                              // 64 floats
  _Float16* wp = (_Float16*)((char*)d_ws + 256);         // 202*512 halves

  invert_q_kernel<<<1, 64, 0, stream>>>(Q, Qi);
  pack_weights_kernel<<<NTILES_TOTAL, 256, 0, stream>>>(W1, W2, W3, W4, wp);
  optnet_kernel<<<NBLOCKS, 256, SMEM_TOTAL, stream>>>(x, Qi, wp, W0, b0, b1, b2, b3, b4, out);
}
